// FullModel_28819230556830
// MI455X (gfx1250) — matrix operand order repaired and checked by reading, not compiled, not run on hardware
//
#include <hip/hip_runtime.h>
#include <math.h>

// ---------------------------------------------------------------------------
// Hyperbolic TimeLSTM encoder + LSTM decoder for MI455X (gfx1250, wave32).
// Matmuls use v_wmma_f32_16x16x32_bf16 (16x16x32, f32 accumulate).
// v2: 2 M-tiles per wave in the wide GEMMs -> each B (weight) fragment feeds
//     two WMMAs, halving L1/L2 weight traffic (weights live in the 192MB L2).
// ---------------------------------------------------------------------------

typedef __bf16 bf16;
typedef __attribute__((ext_vector_type(16))) __bf16 v16bf;
typedef __attribute__((ext_vector_type(8)))  float  v8f;
typedef __attribute__((ext_vector_type(4)))  float  vf4;

#define EPSF 1e-15f
#define MAXN 0.99999f   // 1 - 1e-5

static constexpr int Bn = 1024;   // batch
static constexpr int Tn = 75;     // time steps
static constexpr int IN = 768;    // input dim
static constexpr int Hn = 512;    // hidden
static constexpr int Gn = 2048;   // 4*H

// ---------------------------------------------------------------------------
// WMMA fragment loaders (CDNA5 ISA 7.12.2 layouts)
// A 16x32 bf16: lane m(0..15)/m+16 holds row m; lo-lane K:{0..7,16..23},
//               hi-lane K:{8..15,24..31} -> two contiguous 8-elt chunks.
// B 32x16 bf16: lane holds column (lane&15); same K split as A: lo-lane
//               K:{0..7,16..23}, hi-lane K:{8..15,24..31} (W is [N,K]).
// C/D 16x16 f32: VGPR r -> M = r + 8*(lane>>4), N = lane&15.
// ---------------------------------------------------------------------------

__device__ __forceinline__ v16bf load_a_bf16(const bf16* base, int lda, int m,
                                             int k, int lane) {
  int hi = lane >> 4;
  const bf16* p = base + (size_t)m * lda + k + hi * 8;
  union { v16bf v; vf4 q[2]; } u;
  u.q[0] = *(const vf4*)(p);
  u.q[1] = *(const vf4*)(p + 16);
  return u.v;
}

__device__ __forceinline__ v16bf load_a_f32s(const float* base, int lda, float s,
                                             int m, int k, int lane) {
  int hi = lane >> 4;
  const float* p = base + (size_t)m * lda + k + hi * 8;
  union { v16bf v; bf16 e[16]; } u;
#pragma unroll
  for (int j = 0; j < 8; ++j) {
    u.e[j]     = (bf16)(p[j] * s);
    u.e[8 + j] = (bf16)(p[16 + j] * s);
  }
  return u.v;
}

__device__ __forceinline__ v16bf load_b_frag(const bf16* W, int ldb, int n0,
                                             int k, int lane) {
  int col = lane & 15;
  int kb  = (lane >> 4) * 8;
  const bf16* p = W + (size_t)(n0 + col) * ldb + k + kb;
  union { v16bf v; vf4 q[2]; } u;
  u.q[0] = *(const vf4*)(p);
  u.q[1] = *(const vf4*)(p + 16);
  return u.v;
}

// A-operand descriptor: either bf16 row-major, or f32 row-major with an
// optional per-row scale (used for the logmap0/proj/expmap0-collapsed input).
struct ASrc {
  const bf16*  bf;
  const float* f32;
  const float* rscale;
  int rs_stride, rs_off;
  int lda, K;
};

template <int MT>
__device__ __forceinline__ void run_operand(const ASrc a, const bf16* W, int n0,
                                            int lane, int m0, v8f acc[MT][4]) {
  int mrow = m0 + (lane & 15);
  if (a.bf) {
    for (int k = 0; k < a.K; k += 32) {
      v16bf af[MT];
#pragma unroll
      for (int mt = 0; mt < MT; ++mt)
        af[mt] = load_a_bf16(a.bf, a.lda, mrow + 16 * mt, k, lane);
#pragma unroll
      for (int nt = 0; nt < 4; ++nt) {
        v16bf bfr = load_b_frag(W, a.K, n0 + nt * 16, k, lane);
#pragma unroll
        for (int mt = 0; mt < MT; ++mt)
          acc[mt][nt] = __builtin_amdgcn_wmma_f32_16x16x32_bf16(
              false, af[mt], false, bfr, (short)0, acc[mt][nt], false, false);
      }
    }
  } else {
    float s[MT];
#pragma unroll
    for (int mt = 0; mt < MT; ++mt)
      s[mt] = a.rscale
                  ? a.rscale[(size_t)(mrow + 16 * mt) * a.rs_stride + a.rs_off]
                  : 1.f;
    for (int k = 0; k < a.K; k += 32) {
      v16bf af[MT];
#pragma unroll
      for (int mt = 0; mt < MT; ++mt)
        af[mt] = load_a_f32s(a.f32, a.lda, s[mt], mrow + 16 * mt, k, lane);
#pragma unroll
      for (int nt = 0; nt < 4; ++nt) {
        v16bf bfr = load_b_frag(W, a.K, n0 + nt * 16, k, lane);
#pragma unroll
        for (int mt = 0; mt < MT; ++mt)
          acc[mt][nt] = __builtin_amdgcn_wmma_f32_16x16x32_bf16(
              false, af[mt], false, bfr, (short)0, acc[mt][nt], false, false);
      }
    }
  }
}

// C[M,N] = act( A1 @ W1^T (+ A2 @ W2^T) + bias1 (+ bias2) )
// M fixed = Bn. Block = 128 threads (4 waves); wave w covers MT 16-row tiles;
// block covers 64 columns (4 WMMA tiles, reused by all M-tiles of a wave).
// OP: 0=none 1=sigmoid 2=tanh 3=relu.  MT: M-tiles per wave (1 or 2).
template <int OP, int MT>
__global__ void __launch_bounds__(128)
gemm_wmma(ASrc a1, ASrc a2, const bf16* W1, const bf16* W2,
          const float* bias1, const float* bias2,
          float* C, bf16* Cbf, int N) {
  int lane = threadIdx.x & 31;
  int wave = threadIdx.x >> 5;
  int m0 = blockIdx.y * (64 * MT) + wave * (16 * MT);
  int n0 = blockIdx.x * 64;

  v8f acc[MT][4];
#pragma unroll
  for (int mt = 0; mt < MT; ++mt)
#pragma unroll
    for (int nt = 0; nt < 4; ++nt)
#pragma unroll
      for (int r = 0; r < 8; ++r) acc[mt][nt][r] = 0.f;

  run_operand<MT>(a1, W1, n0, lane, m0, acc);
  if (W2) run_operand<MT>(a2, W2, n0, lane, m0, acc);

  int hi = lane >> 4, col = lane & 15;
#pragma unroll
  for (int nt = 0; nt < 4; ++nt) {
    int n = n0 + nt * 16 + col;
    float bia = (bias1 ? bias1[n] : 0.f) + (bias2 ? bias2[n] : 0.f);
#pragma unroll
    for (int mt = 0; mt < MT; ++mt) {
#pragma unroll
      for (int r = 0; r < 8; ++r) {
        int m = m0 + 16 * mt + r + hi * 8;
        float v = acc[mt][nt][r] + bia;
        if (OP == 1) v = 1.f / (1.f + __expf(-v));
        else if (OP == 2) v = tanhf(v);
        else if (OP == 3) v = fmaxf(v, 0.f);
        if (C)   C[(size_t)m * N + n] = v;
        if (Cbf) Cbf[(size_t)m * N + n] = (bf16)v;
      }
    }
  }
}

// ---------------------------------------------------------------------------
// Elementwise / reduction kernels
// ---------------------------------------------------------------------------

__global__ void cvt_bf16_kernel(const float* src, bf16* dst, int n) {
  int i = blockIdx.x * blockDim.x + threadIdx.x;
  if (i < n) dst[i] = (bf16)src[i];
}

__global__ void zero_f32_kernel(float* p, int n) {
  int i = blockIdx.x * blockDim.x + threadIdx.x;
  if (i < n) p[i] = 0.f;
}

__global__ void zero_bf16_kernel(bf16* p, int n) {
  int i = blockIdx.x * blockDim.x + threadIdx.x;
  if (i < n) p[i] = (bf16)0.f;
}

// Per-row scale of logmap0(proj(expmap0(x))): one block per (b,t) row of len D
__global__ void xscale_kernel(const float* x, float* xscale, int D) {
  __shared__ float red[256];
  int row = blockIdx.x, tid = threadIdx.x;
  const float* p = x + (size_t)row * D;
  float s = 0.f;
  for (int j = tid; j < D; j += 256) { float v = p[j]; s += v * v; }
  red[tid] = s; __syncthreads();
  for (int off = 128; off > 0; off >>= 1) {
    if (tid < off) red[tid] += red[tid + off];
    __syncthreads();
  }
  if (tid == 0) {
    float n1 = fmaxf(sqrtf(red[0]), EPSF);
    float t1 = tanhf(n1);
    float s1 = t1 / n1;                       // expmap0
    if (t1 > MAXN) s1 *= MAXN / t1;           // proj
    float n3 = fmaxf(fminf(t1, MAXN), EPSF);  // norm after proj
    float l = atanhf(fminf(n3, MAXN)) / n3;   // logmap0
    xscale[row] = s1 * l;
  }
}

// t_s[t][b] from tf = proj(expmap0(time_feats)), norm taken over T axis
__global__ void time_kernel(const float* tfeat, float* ts) {
  __shared__ float red[128];
  __shared__ float ssc;
  int b = blockIdx.x, tid = threadIdx.x;
  const float* p = tfeat + (size_t)b * Tn;
  float s = 0.f;
  for (int j = tid; j < Tn; j += 128) { float v = p[j]; s += v * v; }
  red[tid] = s; __syncthreads();
  for (int off = 64; off > 0; off >>= 1) {
    if (tid < off) red[tid] += red[tid + off];
    __syncthreads();
  }
  if (tid == 0) {
    float n = fmaxf(sqrtf(red[0]), EPSF);
    float t1 = tanhf(n);
    float sc = t1 / n;
    if (t1 > MAXN) sc *= MAXN / t1;
    ssc = sc;
  }
  __syncthreads();
  for (int j = tid; j < Tn; j += 128) ts[(size_t)j * Bn + b] = p[j] * ssc;
}

// tangent-space states: ht=logmap0(h), ct=logmap0(c); one block per batch row
__global__ void make_tangent_kernel(const float* h, const float* c,
                                    bf16* ht_bf, bf16* ct_bf, float* ct_f) {
  __shared__ float red[256];
  __shared__ float ssc;
  int b = blockIdx.x, tid = threadIdx.x;
  const float* ph = h + (size_t)b * Hn;
  float h0 = ph[tid], h1 = ph[tid + 256];
  red[tid] = h0 * h0 + h1 * h1; __syncthreads();
  for (int off = 128; off > 0; off >>= 1) {
    if (tid < off) red[tid] += red[tid + off];
    __syncthreads();
  }
  if (tid == 0) {
    float n = fmaxf(sqrtf(red[0]), EPSF);
    ssc = atanhf(fminf(n, MAXN)) / n;
  }
  __syncthreads();
  ht_bf[(size_t)b * Hn + tid]       = (bf16)(h0 * ssc);
  ht_bf[(size_t)b * Hn + tid + 256] = (bf16)(h1 * ssc);
  __syncthreads();

  const float* pc = c + (size_t)b * Hn;
  float c0 = pc[tid], c1 = pc[tid + 256];
  red[tid] = c0 * c0 + c1 * c1; __syncthreads();
  for (int off = 128; off > 0; off >>= 1) {
    if (tid < off) red[tid] += red[tid + off];
    __syncthreads();
  }
  if (tid == 0) {
    float n = fmaxf(sqrtf(red[0]), EPSF);
    ssc = atanhf(fminf(n, MAXN)) / n;
  }
  __syncthreads();
  float cv0 = c0 * ssc, cv1 = c1 * ssc;
  ct_bf[(size_t)b * Hn + tid]       = (bf16)cv0;
  ct_bf[(size_t)b * Hn + tid + 256] = (bf16)cv1;
  ct_f[(size_t)b * Hn + tid]        = cv0;
  ct_f[(size_t)b * Hn + tid + 256]  = cv1;
}

// TimeLSTM gate combine + proj(expmap0()) back onto the ball (in place h,c)
__global__ void enc_combine_kernel(const float* gates, const float* cs1,
                                   const float* ct_f, const float* ts, int t,
                                   float* h, float* c) {
  __shared__ float red[256];
  __shared__ float ssc;
  int b = blockIdx.x, tid = threadIdx.x;
  float tsb = ts[(size_t)t * Bn + b];
  const float* g = gates + (size_t)b * Gn;
  float cn[2], hn[2];
#pragma unroll
  for (int u = 0; u < 2; ++u) {
    int j = tid + u * 256;
    float ctv  = ct_f[(size_t)b * Hn + j];
    float c1   = cs1[(size_t)b * Hn + j];
    float cadj = ctv - c1 + c1 * tsb;
    float f    = g[j];
    float i    = g[Hn + j];
    float o    = g[2 * Hn + j];
    float ctmp = g[3 * Hn + j];
    cn[u] = f * cadj + i * ctmp;
    hn[u] = o * tanhf(cn[u]);
  }
  // c: proj(expmap0(c_new))
  red[tid] = cn[0] * cn[0] + cn[1] * cn[1]; __syncthreads();
  for (int off = 128; off > 0; off >>= 1) {
    if (tid < off) red[tid] += red[tid + off];
    __syncthreads();
  }
  if (tid == 0) {
    float n = fmaxf(sqrtf(red[0]), EPSF);
    float t1 = tanhf(n);
    float s = t1 / n;
    if (t1 > MAXN) s *= MAXN / t1;
    ssc = s;
  }
  __syncthreads();
  c[(size_t)b * Hn + tid]       = cn[0] * ssc;
  c[(size_t)b * Hn + tid + 256] = cn[1] * ssc;
  __syncthreads();
  // h: proj(expmap0(h_new))
  red[tid] = hn[0] * hn[0] + hn[1] * hn[1]; __syncthreads();
  for (int off = 128; off > 0; off >>= 1) {
    if (tid < off) red[tid] += red[tid + off];
    __syncthreads();
  }
  if (tid == 0) {
    float n = fmaxf(sqrtf(red[0]), EPSF);
    float t1 = tanhf(n);
    float s = t1 / n;
    if (t1 > MAXN) s *= MAXN / t1;
    ssc = s;
  }
  __syncthreads();
  h[(size_t)b * Hn + tid]       = hn[0] * ssc;
  h[(size_t)b * Hn + tid + 256] = hn[1] * ssc;
}

// hx = proj(logmap0(h_enc)) in place, plus bf16 copy of h; same for c
__global__ void final_tangent_kernel(float* h, float* c, bf16* h_bf) {
  __shared__ float red[256];
  __shared__ float ssc;
  int b = blockIdx.x, tid = threadIdx.x;
  float* ph = h + (size_t)b * Hn;
  float h0 = ph[tid], h1 = ph[tid + 256];
  red[tid] = h0 * h0 + h1 * h1; __syncthreads();
  for (int off = 128; off > 0; off >>= 1) {
    if (tid < off) red[tid] += red[tid + off];
    __syncthreads();
  }
  if (tid == 0) {
    float n = fmaxf(sqrtf(red[0]), EPSF);
    float l = atanhf(fminf(n, MAXN)) / n;
    float ny = l * n;
    float s = l;
    if (ny > MAXN) s *= MAXN / ny;
    ssc = s;
  }
  __syncthreads();
  float v0 = h0 * ssc, v1 = h1 * ssc;
  ph[tid] = v0; ph[tid + 256] = v1;
  h_bf[(size_t)b * Hn + tid]       = (bf16)v0;
  h_bf[(size_t)b * Hn + tid + 256] = (bf16)v1;
  __syncthreads();

  float* pc = c + (size_t)b * Hn;
  float c0 = pc[tid], c1 = pc[tid + 256];
  red[tid] = c0 * c0 + c1 * c1; __syncthreads();
  for (int off = 128; off > 0; off >>= 1) {
    if (tid < off) red[tid] += red[tid + off];
    __syncthreads();
  }
  if (tid == 0) {
    float n = fmaxf(sqrtf(red[0]), EPSF);
    float l = atanhf(fminf(n, MAXN)) / n;
    float ny = l * n;
    float s = l;
    if (ny > MAXN) s *= MAXN / ny;
    ssc = s;
  }
  __syncthreads();
  pc[tid] = c0 * ssc;
  pc[tid + 256] = c1 * ssc;
}

__device__ __forceinline__ float sigm(float x) { return 1.f / (1.f + __expf(-x)); }

// torch LSTMCell combine (gate order i,f,g,o); writes h (f32 + bf16), c
__global__ void dec_combine_kernel(const float* gates, float* c, float* h,
                                   bf16* h_bf) {
  int idx = blockIdx.x * blockDim.x + threadIdx.x;
  if (idx >= Bn * Hn) return;
  int b = idx >> 9, j = idx & (Hn - 1);
  const float* g = gates + (size_t)b * Gn;
  float i_ = g[j], f_ = g[Hn + j], gg = g[2 * Hn + j], o_ = g[3 * Hn + j];
  float cc = sigm(f_) * c[idx] + sigm(i_) * tanhf(gg);
  float hh = sigm(o_) * tanhf(cc);
  c[idx] = cc;
  h[idx] = hh;
  h_bf[idx] = (bf16)hh;
}

// num_spans = softmax(hx @ W_span^T + b_span)  (N = 4)
__global__ void span_kernel(const float* hx, const float* W, const float* bia,
                            float* out) {
  int b = blockIdx.x * blockDim.x + threadIdx.x;
  if (b >= Bn) return;
  float a[4];
  float mx = -1e30f;
#pragma unroll
  for (int n = 0; n < 4; ++n) {
    float s = bia[n];
    const float* hr = hx + (size_t)b * Hn;
    const float* wr = W + (size_t)n * Hn;
    for (int k = 0; k < Hn; ++k) s += hr[k] * wr[k];
    a[n] = s;
    mx = fmaxf(mx, s);
  }
  float den = 0.f;
#pragma unroll
  for (int n = 0; n < 4; ++n) { a[n] = __expf(a[n] - mx); den += a[n]; }
#pragma unroll
  for (int n = 0; n < 4; ++n) out[(size_t)b * 4 + n] = a[n] / den;
}

// out_step = softmax(h @ W_fc_out^T + b_fc_out)  (N = 2), scattered (B,10,2)
__global__ void fcout_kernel(const float* h, const float* W, const float* bia,
                             float* out, int step) {
  int b = blockIdx.x * blockDim.x + threadIdx.x;
  if (b >= Bn) return;
  float a[2];
#pragma unroll
  for (int n = 0; n < 2; ++n) {
    float s = bia[n];
    const float* hr = h + (size_t)b * Hn;
    const float* wr = W + (size_t)n * Hn;
    for (int k = 0; k < Hn; ++k) s += hr[k] * wr[k];
    a[n] = s;
  }
  float mx = fmaxf(a[0], a[1]);
  float e0 = __expf(a[0] - mx), e1 = __expf(a[1] - mx);
  float den = e0 + e1;
  out[(size_t)b * 20 + step * 2 + 0] = e0 / den;
  out[(size_t)b * 20 + step * 2 + 1] = e1 / den;
}

// ---------------------------------------------------------------------------
// Host driver
// ---------------------------------------------------------------------------

static ASrc abf(const bf16* p, int lda, int K) {
  ASrc a; a.bf = p; a.f32 = nullptr; a.rscale = nullptr;
  a.rs_stride = 0; a.rs_off = 0; a.lda = lda; a.K = K; return a;
}
static ASrc af32(const float* p, const float* rs, int rss, int rso, int lda, int K) {
  ASrc a; a.bf = nullptr; a.f32 = p; a.rscale = rs;
  a.rs_stride = rss; a.rs_off = rso; a.lda = lda; a.K = K; return a;
}

extern "C" void kernel_launch(void* const* d_in, const int* in_sizes, int n_in,
                              void* d_out, int out_size, void* d_ws, size_t ws_size,
                              hipStream_t stream) {
  const float* x       = (const float*)d_in[0];
  const float* tfeat   = (const float*)d_in[1];
  const float* Wd      = (const float*)d_in[2];
  const float* bd      = (const float*)d_in[3];
  const float* W_all   = (const float*)d_in[4];
  const float* b_all   = (const float*)d_in[5];
  const float* U_all   = (const float*)d_in[6];
  const float* b_U     = (const float*)d_in[7];
  const float* W_ih    = (const float*)d_in[8];
  const float* b_ih    = (const float*)d_in[9];
  const float* W_hh    = (const float*)d_in[10];
  const float* b_hh    = (const float*)d_in[11];
  const float* W_fc_in = (const float*)d_in[12];
  const float* b_fc_in = (const float*)d_in[13];
  const float* W_fc_out= (const float*)d_in[14];
  const float* b_fc_out= (const float*)d_in[15];
  const float* W_span  = (const float*)d_in[16];
  const float* b_span  = (const float*)d_in[17];
  float* out = (float*)d_out;

  char* ws = (char*)d_ws;
  auto alloc = [&](size_t bytes) -> char* {
    char* p = ws;
    ws += (bytes + 255) & ~(size_t)255;
    return p;
  };

  bf16* Wd_bf    = (bf16*)alloc((size_t)Hn * Hn * 2);
  bf16* Wall_bf  = (bf16*)alloc((size_t)Gn * Hn * 2);
  bf16* Uall_bf  = (bf16*)alloc((size_t)Gn * IN * 2);
  bf16* Wih_bf   = (bf16*)alloc((size_t)Gn * IN * 2);
  bf16* Whh_bf   = (bf16*)alloc((size_t)Gn * Hn * 2);
  bf16* Wfcin_bf = (bf16*)alloc((size_t)IN * Hn * 2);
  float* xscale  = (float*)alloc((size_t)Bn * Tn * 4);
  float* ts      = (float*)alloc((size_t)Tn * Bn * 4);
  float* h       = (float*)alloc((size_t)Bn * Hn * 4);
  float* c       = (float*)alloc((size_t)Bn * Hn * 4);
  bf16* ht_bf    = (bf16*)alloc((size_t)Bn * Hn * 2);
  bf16* ct_bf    = (bf16*)alloc((size_t)Bn * Hn * 2);
  float* ct_f    = (float*)alloc((size_t)Bn * Hn * 4);
  float* cs1     = (float*)alloc((size_t)Bn * Hn * 4);
  float* gates   = (float*)alloc((size_t)Bn * Gn * 4);
  bf16* h_bf     = (bf16*)alloc((size_t)Bn * Hn * 2);
  bf16* inp_bf   = (bf16*)alloc((size_t)Bn * IN * 2);
  (void)ws_size; (void)in_sizes; (void)n_in; (void)out_size;

  auto cvt = [&](const float* s, bf16* d, int n) {
    cvt_bf16_kernel<<<(n + 255) / 256, 256, 0, stream>>>(s, d, n);
  };
  cvt(Wd, Wd_bf, Hn * Hn);
  cvt(W_all, Wall_bf, Gn * Hn);
  cvt(U_all, Uall_bf, Gn * IN);
  cvt(W_ih, Wih_bf, Gn * IN);
  cvt(W_hh, Whh_bf, Gn * Hn);
  cvt(W_fc_in, Wfcin_bf, IN * Hn);

  xscale_kernel<<<Bn * Tn, 256, 0, stream>>>(x, xscale, IN);
  time_kernel<<<Bn, 128, 0, stream>>>(tfeat, ts);
  zero_f32_kernel<<<(Bn * Hn + 255) / 256, 256, 0, stream>>>(h, Bn * Hn);
  zero_f32_kernel<<<(Bn * Hn + 255) / 256, 256, 0, stream>>>(c, Bn * Hn);

  ASrc none = abf(nullptr, 0, 0);

  // ---------------- encoder: 75 serialized TimeLSTM steps ----------------
  for (int t = 0; t < Tn; ++t) {
    make_tangent_kernel<<<Bn, 256, 0, stream>>>(h, c, ht_bf, ct_bf, ct_f);

    // c_s1 = tanh(ct @ Wd^T + bd)    (1024 x 512, K = 512)  MT=1
    gemm_wmma<2, 1><<<dim3(Hn / 64, Bn / 64), 128, 0, stream>>>(
        abf(ct_bf, Hn, Hn), none, Wd_bf, nullptr, bd, nullptr, cs1, nullptr, Hn);

    // gates = sigmoid(ht @ W_all^T + xt @ U_all^T + b_all + b_U)  MT=2
    // xt is generated on the fly: bf16(x[b,t,:] * xscale[b*T + t])
    gemm_wmma<1, 2><<<dim3(Gn / 64, Bn / 128), 128, 0, stream>>>(
        abf(ht_bf, Hn, Hn),
        af32(x + (size_t)t * IN, xscale, Tn, t, Tn * IN, IN),
        Wall_bf, Uall_bf, b_all, b_U, gates, nullptr, Gn);

    enc_combine_kernel<<<Bn, 256, 0, stream>>>(gates, cs1, ct_f, ts, t, h, c);
  }

  // hx = proj(logmap0(h_enc)), cx = proj(logmap0(c_enc)); bf16 copy of hx
  final_tangent_kernel<<<Bn, 256, 0, stream>>>(h, c, h_bf);

  // num_spans = softmax(hx @ W_span^T + b_span) -> out[0 : B*4]
  span_kernel<<<(Bn + 255) / 256, 256, 0, stream>>>(h, W_span, b_span, out);

  zero_bf16_kernel<<<(Bn * IN + 255) / 256, 256, 0, stream>>>(inp_bf, Bn * IN);

  // ---------------- decoder: 10 LSTM steps ----------------
  for (int step = 0; step < 10; ++step) {
    // gates = inp @ W_ih^T + h @ W_hh^T + b_ih + b_hh   (raw)  MT=2
    gemm_wmma<0, 2><<<dim3(Gn / 64, Bn / 128), 128, 0, stream>>>(
        abf(inp_bf, IN, IN), abf(h_bf, Hn, Hn),
        Wih_bf, Whh_bf, b_ih, b_hh, gates, nullptr, Gn);

    dec_combine_kernel<<<(Bn * Hn + 255) / 256, 256, 0, stream>>>(gates, c, h, h_bf);

    // inp = relu(h @ W_fc_in^T + b_fc_in)  -> bf16 directly  MT=1
    gemm_wmma<3, 1><<<dim3(IN / 64, Bn / 64), 128, 0, stream>>>(
        abf(h_bf, Hn, Hn), none, Wfcin_bf, nullptr, b_fc_in, nullptr,
        nullptr, inp_bf, IN);

    // out_step = softmax(h @ W_fc_out^T + b_fc_out) -> out[B*4 + ...]
    fcout_kernel<<<(Bn + 255) / 256, 256, 0, stream>>>(
        h, W_fc_out, b_fc_out, out + (size_t)Bn * 4, step);
  }
}
